// AdaptiveQuantizedLinear_64330020159899
// MI455X (gfx1250) — matrix operand order repaired and checked by reading, not compiled, not run on hardware
//
#include <hip/hip_runtime.h>
#include <hip/hip_bf16.h>

typedef __attribute__((ext_vector_type(16))) __bf16 v16bf;
typedef __attribute__((ext_vector_type(8)))  float  v8f;
typedef unsigned int u32;

#define MDIM 8192        // B*S
#define NDIM 11008       // O
#define KDIM 4096        // I
#define NGRP 64          // I / G

__device__ __forceinline__ u32 pack2bf16(float lo, float hi) {
  // round-to-nearest-even fp32 -> bf16, packed pair (pure integer ops)
  u32 a = __float_as_uint(lo);
  u32 b = __float_as_uint(hi);
  a += 0x7FFFu + ((a >> 16) & 1u);
  b += 0x7FFFu + ((b >> 16) & 1u);
  return (a >> 16) | (b & 0xFFFF0000u);
}

union Frag {
  v16bf v;
  uint4 q[2];
};

// ============================================================================
// Pre-pass 1: x fp32 -> bf16 (each thread: 8 elements)
// ============================================================================
__global__ __launch_bounds__(256)
void convert_x_kernel(const float* __restrict__ x, u32* __restrict__ xb) {
  const size_t gid = (size_t)blockIdx.x * 256 + threadIdx.x;
  const float4* src = (const float4*)(x + gid * 8);
  float4 a = src[0], b = src[1];
  uint4 p;
  p.x = pack2bf16(a.x, a.y);
  p.y = pack2bf16(a.z, a.w);
  p.z = pack2bf16(b.x, b.y);
  p.w = pack2bf16(b.z, b.w);
  *(uint4*)(xb + gid * 4) = p;
}

// ============================================================================
// Pre-pass 2: W dequant (4-bit codes, group affine, row scale, mask) -> bf16
// Each thread: 8 elements of one row (group-uniform since 8 | 64).
// ============================================================================
__global__ __launch_bounds__(256)
void dequant_w_kernel(const int* __restrict__ Wq,
                      const unsigned char* __restrict__ mask,
                      const float* __restrict__ scales,
                      const float* __restrict__ zeros,
                      const float* __restrict__ scale2,
                      u32* __restrict__ wb) {
  const size_t gid = (size_t)blockIdx.x * 256 + threadIdx.x;
  const int n  = (int)(gid >> 9);            // 512 threads per row (4096/8)
  const int kk = ((int)gid & 511) << 3;
  const int g  = kk >> 6;
  const float sc = scales[(size_t)n * NGRP + g] * scale2[n];
  const float nz = -zeros[(size_t)n * NGRP + g] * sc;   // dequant = fma(c, sc, nz)
  const int4* cp = (const int4*)(Wq + (size_t)n * KDIM + kk);
  int4 c0 = cp[0], c1 = cp[1];
  uint2 mq = *(const uint2*)(mask + (size_t)n * KDIM + kk);
  int cc[8]  = {c0.x, c0.y, c0.z, c0.w, c1.x, c1.y, c1.z, c1.w};
  u32 mm[2]  = {mq.x, mq.y};
  float w[8];
#pragma unroll
  for (int j = 0; j < 8; ++j) {
    const u32 mb = (mm[j >> 2] >> ((j & 3) * 8)) & 0xFFu;
    w[j] = mb ? fmaf((float)cc[j], sc, nz) : 0.0f;
  }
  uint4 p;
  p.x = pack2bf16(w[0], w[1]);
  p.y = pack2bf16(w[2], w[3]);
  p.z = pack2bf16(w[4], w[5]);
  p.w = pack2bf16(w[6], w[7]);
  *(uint4*)(wb + ((size_t)n * KDIM + kk) / 2) = p;
}

// ============================================================================
// Main GEMM: bf16 x bf16 -> fp32, 256(M) x 128(N) block, BK=64, async-to-LDS
// ============================================================================
#define GBM 256
#define GBN 128
#define GBK 64
#define ASTRIDE 144                 // 64 bf16 data + 8 bf16 pad (bank spread)
#define ABUF (GBM * ASTRIDE)        // 36864 B per stage
#define BBUF (GBN * ASTRIDE)        // 18432 B per stage
#define NSTAGE (KDIM / GBK)         // 64

// async copy: 16B per lane; INST_OFFSET applies to BOTH lds and global address
#define ASYNC_CP(lds, voff, base, imm)                                    \
  asm volatile("global_load_async_to_lds_b128 %0, %1, %2 offset:" #imm    \
               :: "v"(lds), "v"(voff), "s"(base) : "memory")

__global__ __launch_bounds__(256)
void gemm_bf16_async_kernel(const unsigned char* __restrict__ xb,
                            const unsigned char* __restrict__ wb,
                            const float* __restrict__ bias,
                            float* __restrict__ out)
{
  __shared__ __align__(16) unsigned char Asm2[2][ABUF];
  __shared__ __align__(16) unsigned char Bsm2[2][BBUF];

  const int t   = threadIdx.x;
  const int bn0 = blockIdx.x * GBN;
  const int bm0 = blockIdx.y * GBM;

  // -------- per-thread async staging addresses --------
  // A: thread t owns tile row t (256 rows), 8 chunks of 16B (128B of k)
  // B: thread t owns tile row t>>1 (128 rows), k-half t&1, 4 chunks of 16B
  const u32 aLds0 = (u32)(uintptr_t)&Asm2[0][0] + (u32)t * ASTRIDE;
  const u32 bLds0 = (u32)(uintptr_t)&Bsm2[0][0] + (u32)(t >> 1) * ASTRIDE + (u32)(t & 1) * 64;
  const u32 aVoff0 = (u32)(bm0 + t) * (KDIM * 2);
  const u32 bVoff0 = (u32)(bn0 + (t >> 1)) * (KDIM * 2) + (u32)(t & 1) * 64;

  auto issue_stage = [&](int kt, int buf) {
    const u32 kbB = (u32)kt * (GBK * 2);
    const u32 av  = aVoff0 + kbB;
    const u32 bv  = bVoff0 + kbB;
    const u32 al  = aLds0 + (u32)buf * ABUF;
    const u32 bl  = bLds0 + (u32)buf * BBUF;
    ASYNC_CP(al, av, xb, 0);
    ASYNC_CP(al, av, xb, 16);
    ASYNC_CP(al, av, xb, 32);
    ASYNC_CP(al, av, xb, 48);
    ASYNC_CP(al, av, xb, 64);
    ASYNC_CP(al, av, xb, 80);
    ASYNC_CP(al, av, xb, 96);
    ASYNC_CP(al, av, xb, 112);
    ASYNC_CP(bl, bv, wb, 0);
    ASYNC_CP(bl, bv, wb, 16);
    ASYNC_CP(bl, bv, wb, 32);
    ASYNC_CP(bl, bv, wb, 48);
  };

  // -------- wave tiling: 8 waves = 2(M) x 4(N); wave -> 128x32 = 8x2 tiles ----
  const int wid   = t >> 5;
  const int lane  = t & 31;
  const int waveM = wid >> 2;   // 0..1
  const int waveN = wid & 3;    // 0..3
  const int l15   = lane & 15;
  const int lh    = lane >> 4;

  v8f acc[8][2] = {};

  issue_stage(0, 0);

  for (int kt = 0; kt < NSTAGE; ++kt) {
    asm volatile("s_wait_asynccnt 0x0" ::: "memory");   // my async writes landed
    __syncthreads();                                    // everyone's landed
    const int buf = kt & 1;
    if (kt + 1 < NSTAGE) issue_stage(kt + 1, buf ^ 1);

    const unsigned char* Ab = &Asm2[buf][0];
    const unsigned char* Bb = &Bsm2[buf][0];

#pragma unroll
    for (int ks = 0; ks < 2; ++ks) {
      Frag b0, b1;
      {
        // B (32x16): lane -> col n=l15, k = ks*32 + lh*16 + 0..15 (32B run)
        const int rowB = waveN * 32 + l15;
        const u32 off0 = (u32)rowB * ASTRIDE + (u32)ks * 64 + (u32)lh * 32;
        b0.q[0] = *(const uint4*)(Bb + off0);
        b0.q[1] = *(const uint4*)(Bb + off0 + 16);
        const u32 off1 = off0 + 16u * ASTRIDE;
        b1.q[0] = *(const uint4*)(Bb + off1);
        b1.q[1] = *(const uint4*)(Bb + off1 + 16);
      }
#pragma unroll
      for (int mt = 0; mt < 8; ++mt) {
        // A (16x32): lane -> row m=l15, k = ks*32 + lh*16 + 0..15 (32B run)
        Frag a;
        const int rowA = waveM * 128 + mt * 16 + l15;
        const u32 off  = (u32)rowA * ASTRIDE + (u32)ks * 64 + (u32)lh * 32;
        a.q[0] = *(const uint4*)(Ab + off);
        a.q[1] = *(const uint4*)(Ab + off + 16);
        acc[mt][0] = __builtin_amdgcn_wmma_f32_16x16x32_bf16(
            false, a.v, false, b0.v, (short)0, acc[mt][0], false, false);
        acc[mt][1] = __builtin_amdgcn_wmma_f32_16x16x32_bf16(
            false, a.v, false, b1.v, (short)0, acc[mt][1], false, false);
      }
    }
  }

  // -------- epilogue: bias add + store (VGPR r -> M=r+8*lh, lane -> N) --------
#pragma unroll
  for (int nt = 0; nt < 2; ++nt) {
    const int n = bn0 + waveN * 32 + nt * 16 + l15;
    const float bv = bias[n];
#pragma unroll
    for (int mt = 0; mt < 8; ++mt) {
      const int m0 = bm0 + waveM * 128 + mt * 16 + (lh << 3);
#pragma unroll
      for (int rr = 0; rr < 8; ++rr) {
        out[(size_t)(m0 + rr) * NDIM + n] = acc[mt][nt][rr] + bv;
      }
    }
  }
}

// ============================================================================
// Fallback: fully fused kernel (round-1), used when workspace is too small
// ============================================================================
#define BM 128
#define BN 128
#define BK 64
#define SU 36
#define KSTAGES (KDIM / BK)

__global__ __launch_bounds__(256)
void aql_fused_kernel(const float* __restrict__ x,
                      const float* __restrict__ scales,
                      const float* __restrict__ zeros,
                      const float* __restrict__ scale2,
                      const float* __restrict__ bias,
                      const int*   __restrict__ Wq,
                      const unsigned char* __restrict__ mask,
                      float* __restrict__ out)
{
  __shared__ __align__(16) u32 Al[2][BM * SU];
  __shared__ __align__(16) u32 Bl[2][BN * SU];

  const int t   = threadIdx.x;
  const int bn0 = blockIdx.x * BN;
  const int bm0 = blockIdx.y * BM;
  const int r   = t >> 1;
  const int h   = t & 1;
  const int kh  = h * 32;
  const int gm  = bm0 + r;
  const int gn  = bn0 + r;
  const float s2 = scale2[gn];

  auto load_stage = [&](int kt, int buf) {
    const int kb = kt * BK;
    const float4* xr = (const float4*)(x + (size_t)gm * KDIM + kb + kh);
    float4 xv[8];
#pragma unroll
    for (int i = 0; i < 8; ++i) xv[i] = xr[i];
    uint4* arow = (uint4*)&Al[buf][r * SU + h * 16];
#pragma unroll
    for (int i = 0; i < 4; ++i) {
      uint4 p;
      p.x = pack2bf16(xv[2*i].x,   xv[2*i].y);
      p.y = pack2bf16(xv[2*i].z,   xv[2*i].w);
      p.z = pack2bf16(xv[2*i+1].x, xv[2*i+1].y);
      p.w = pack2bf16(xv[2*i+1].z, xv[2*i+1].w);
      arow[i] = p;
    }
    const int g = kb >> 6;
    const float sc = scales[(size_t)gn * NGRP + g] * s2;
    const float zp = zeros [(size_t)gn * NGRP + g];
    const int4*  wr = (const int4*)(Wq   + (size_t)gn * KDIM + kb + kh);
    const uint4* mr = (const uint4*)(mask + (size_t)gn * KDIM + kb + kh);
    uint4 mq0 = mr[0], mq1 = mr[1];
    u32 mu[8] = {mq0.x, mq0.y, mq0.z, mq0.w, mq1.x, mq1.y, mq1.z, mq1.w};
    int4 wv[8];
#pragma unroll
    for (int i = 0; i < 8; ++i) wv[i] = wr[i];
    uint4* brow = (uint4*)&Bl[buf][r * SU + h * 16];
#pragma unroll
    for (int i = 0; i < 4; ++i) {
      float wf[8];
      const int* c = (const int*)&wv[2*i];
#pragma unroll
      for (int j = 0; j < 8; ++j) {
        const int e = i * 8 + j;
        const u32 mb = (mu[e >> 2] >> ((e & 3) * 8)) & 0xFFu;
        const float w = ((float)c[j] - zp) * sc;
        wf[j] = mb ? w : 0.0f;
      }
      uint4 p;
      p.x = pack2bf16(wf[0], wf[1]);
      p.y = pack2bf16(wf[2], wf[3]);
      p.z = pack2bf16(wf[4], wf[5]);
      p.w = pack2bf16(wf[6], wf[7]);
      brow[i] = p;
    }
    if (kt + 1 < KSTAGES) {
      __builtin_prefetch(x    + (size_t)gm * KDIM + (kb + BK) + kh, 0, 1);
      __builtin_prefetch(Wq   + (size_t)gn * KDIM + (kb + BK) + kh, 0, 1);
      __builtin_prefetch(mask + (size_t)gn * KDIM + (kb + BK) + kh, 0, 1);
    }
  };

  const int wid   = t >> 5;
  const int lane  = t & 31;
  const int waveM = wid >> 2;
  const int waveN = wid & 3;
  const int l15   = lane & 15;
  const int lh    = lane >> 4;

  v8f acc[4][2] = {};
  load_stage(0, 0);

  for (int kt = 0; kt < KSTAGES; ++kt) {
    __syncthreads();
    const int buf = kt & 1;
    if (kt + 1 < KSTAGES) load_stage(kt + 1, buf ^ 1);
#pragma unroll
    for (int ks = 0; ks < 2; ++ks) {
      Frag a[4], b[2];
#pragma unroll
      for (int nt = 0; nt < 2; ++nt) {
        const int rowB = waveN * 32 + nt * 16 + l15;
        const u32 off  = rowB * SU + ks * 16 + lh * 8;
        const uint4* p = (const uint4*)&Bl[buf][off];
        b[nt].q[0] = p[0];
        b[nt].q[1] = p[1];
      }
#pragma unroll
      for (int mt = 0; mt < 4; ++mt) {
        const int rowA = waveM * 64 + mt * 16 + l15;
        const u32 off  = rowA * SU + ks * 16 + lh * 8;
        a[mt].q[0] = *(const uint4*)&Al[buf][off];
        a[mt].q[1] = *(const uint4*)&Al[buf][off + 4];
      }
#pragma unroll
      for (int mt = 0; mt < 4; ++mt)
#pragma unroll
        for (int nt = 0; nt < 2; ++nt)
          acc[mt][nt] = __builtin_amdgcn_wmma_f32_16x16x32_bf16(
              false, a[mt].v, false, b[nt].v, (short)0, acc[mt][nt],
              false, false);
    }
  }

#pragma unroll
  for (int nt = 0; nt < 2; ++nt) {
    const int n = bn0 + waveN * 32 + nt * 16 + l15;
    const float bv = bias[n];
#pragma unroll
    for (int mt = 0; mt < 4; ++mt) {
      const int m0 = bm0 + waveM * 64 + mt * 16 + (lh << 3);
#pragma unroll
      for (int rr = 0; rr < 8; ++rr) {
        out[(size_t)(m0 + rr) * NDIM + n] = acc[mt][nt][rr] + bv;
      }
    }
  }
}

// ============================================================================
// Host launcher
// ============================================================================
extern "C" void kernel_launch(void* const* d_in, const int* in_sizes, int n_in,
                              void* d_out, int out_size, void* d_ws, size_t ws_size,
                              hipStream_t stream) {
  (void)in_sizes; (void)n_in; (void)out_size;
  const float* x      = (const float*)d_in[0];
  const float* scales = (const float*)d_in[1];
  const float* zeros  = (const float*)d_in[2];
  const float* scale2 = (const float*)d_in[3];
  const float* bias   = (const float*)d_in[4];
  const int*   Wq     = (const int*)d_in[5];
  const unsigned char* mask = (const unsigned char*)d_in[6];
  float* out = (float*)d_out;

  const size_t xb_bytes = (size_t)MDIM * KDIM * 2;   // 67,108,864
  const size_t wb_bytes = (size_t)NDIM * KDIM * 2;   // 90,177,536
  const size_t need = xb_bytes + wb_bytes;           // 157,286,400

  if (ws_size >= need) {
    u32* xb = (u32*)d_ws;
    u32* wb = (u32*)((char*)d_ws + xb_bytes);
    convert_x_kernel<<<dim3((MDIM * KDIM) / (8 * 256)), dim3(256), 0, stream>>>(x, xb);
    dequant_w_kernel<<<dim3((NDIM * KDIM) / (8 * 256)), dim3(256), 0, stream>>>(
        Wq, mask, scales, zeros, scale2, wb);
    gemm_bf16_async_kernel<<<dim3(NDIM / GBN, MDIM / GBM), dim3(256), 0, stream>>>(
        (const unsigned char*)xb, (const unsigned char*)wb, bias, out);
  } else {
    aql_fused_kernel<<<dim3(NDIM / BN, MDIM / BM), dim3(256), 0, stream>>>(
        x, scales, zeros, scale2, bias, Wq, mask, out);
  }
}
